// PointerDecoder_24799141167547
// MI455X (gfx1250) — matrix operand order repaired and checked by reading, not compiled, not run on hardware
//
#include <hip/hip_runtime.h>

// ---------------- problem constants (from reference) ----------------
#define BB     32
#define NN     512
#define TENC   512
#define TDEC   50
#define DEMB_  512
#define DENC_  1024
#define DDEC_  1024
#define NGATE  4096            // 4*DDEC
#define KGATE  1536            // DEMB + DDEC
#define KFEAT  2048            // DDEC + DENC
#define NFEAT  513             // DEMB + 1
#define NEGV   1e20f

typedef __attribute__((ext_vector_type(16))) __bf16 v16bf;
typedef __attribute__((ext_vector_type(8)))  float  v8f;

__device__ __forceinline__ unsigned short f32_to_bf16(float f) {
  unsigned int u = __float_as_uint(f);
  u += 0x7fffu + ((u >> 16) & 1u);       // round-to-nearest-even
  return (unsigned short)(u >> 16);
}

// ---------------------------------------------------------------
// Pack weights W (row-major [N, K] fp32, optionally concat of two
// sources along K) into bf16 WMMA B-fragment order:
//   offset = ((tn*Ktiles + tk)*32 + lane)*16 + i
//   lane: col n = tn*16 + (lane&15);  kbase = (lane>>4)*8
//   i<8 : K = tk*32 + kbase + i ; i>=8 : K = tk*32 + 16 + kbase + (i-8)
// Rows n >= Nrows are zero-padded.
// ---------------------------------------------------------------
__global__ void pack_b_kernel(unsigned short* __restrict__ Bp,
                              const float* __restrict__ W0, int K0,
                              const float* __restrict__ W1, int K1,
                              int Nrows, int Ntiles)
{
  const int K = K0 + K1;
  const int Ktiles = K >> 5;
  const size_t total = (size_t)Ntiles * Ktiles * 512;
  const size_t stride = (size_t)gridDim.x * blockDim.x;
  for (size_t idx = (size_t)blockIdx.x * blockDim.x + threadIdx.x; idx < total; idx += stride) {
    const int i    = (int)(idx & 15);
    const int lane = (int)((idx >> 4) & 31);
    const size_t tlin = idx >> 9;               // tn*Ktiles + tk
    const int tk = (int)(tlin % Ktiles);
    const int tn = (int)(tlin / Ktiles);
    const int n  = tn * 16 + (lane & 15);
    const int k  = tk * 32 + ((i < 8) ? (((lane >> 4) << 3) + i) : (16 + ((lane >> 4) << 3) + (i - 8)));
    float f = 0.0f;
    if (n < Nrows)
      f = (k < K0) ? W0[(size_t)n * K0 + k] : W1[(size_t)n * K1 + (k - K0)];
    Bp[idx] = f32_to_bf16(f);
  }
}

// ---------------------------------------------------------------
// Pack activations X = [S0 | S1] (fp32, 32 rows, K = K0+K1 cols) into
// bf16 WMMA A-fragment order (16-bit 16x32 A layout, ISA 7.12.2):
//   offset = ((r*Ktiles + tk)*32 + lane)*16 + i, r = row-tile (0/1)
//   row M = r*16 + (lane&15); kbase = (lane>>4)*8
//   i<8 : K = tk*32 + kbase + i ; i>=8 : K = tk*32 + 16 + kbase + (i-8)
// ---------------------------------------------------------------
__global__ void pack_a_kernel(unsigned short* __restrict__ Ap,
                              const float* __restrict__ S0, int K0,
                              const float* __restrict__ S1, int K1)
{
  const int K = K0 + K1;
  const int Ktiles = K >> 5;
  const int total = 2 * Ktiles * 512;
  const int stride = gridDim.x * blockDim.x;
  for (int idx = blockIdx.x * blockDim.x + threadIdx.x; idx < total; idx += stride) {
    const int i    = idx & 15;
    const int lane = (idx >> 4) & 31;
    const int tlin = idx >> 9;                  // r*Ktiles + tk
    const int tk = tlin % Ktiles;
    const int r  = tlin / Ktiles;
    const int row = r * 16 + (lane & 15);
    const int kbase = (lane >> 4) << 3;
    const int k = tk * 32 + ((i < 8) ? (kbase + i) : (16 + kbase + (i - 8)));
    const float f = (k < K0) ? S0[(size_t)row * K0 + k]
                             : S1[(size_t)row * K1 + (k - K0)];
    Ap[idx] = f32_to_bf16(f);
  }
}

// ---------------------------------------------------------------
// C[32, Nout] = A[32, K] * W^T + bias   (bf16 in, fp32 accumulate)
// One wave per 16-column tile; each wave runs two WMMA chains (row
// tiles 0-15 and 16-31) over Ktiles iterations of K=32.
// ---------------------------------------------------------------
__global__ void wmma_gemm_kernel(const unsigned short* __restrict__ Ap,
                                 const unsigned short* __restrict__ Bp,
                                 const float* __restrict__ bias,
                                 float* __restrict__ C,
                                 int Ntiles, int Ktiles, int Nout)
{
  const int wave = blockIdx.x * 4 + (threadIdx.x >> 5);  // 4 waves / block
  if (wave >= Ntiles) return;                            // wave-uniform exit
  const int lane = threadIdx.x & 31;
  const int tn = wave;

  v8f c0 = {};
  v8f c1 = {};
  const unsigned short* aBase = Ap + (size_t)lane * 16;
  const unsigned short* bBase = Bp + ((size_t)tn * Ktiles * 32 + lane) * 16;

  for (int tk = 0; tk < Ktiles; ++tk) {
    v16bf a0 = *(const v16bf*)(aBase + (size_t)tk * 512);
    v16bf a1 = *(const v16bf*)(aBase + (size_t)(Ktiles + tk) * 512);
    v16bf b  = *(const v16bf*)(bBase + (size_t)tk * 512);
    c0 = __builtin_amdgcn_wmma_f32_16x16x32_bf16(false, a0, false, b, (short)0, c0, false, false);
    c1 = __builtin_amdgcn_wmma_f32_16x16x32_bf16(false, a1, false, b, (short)0, c1, false, false);
  }

  const int col = tn * 16 + (lane & 15);
  if (col < Nout) {
    const float bv = bias[col];
    const int rowBase = (lane >> 4) << 3;
#pragma unroll
    for (int v = 0; v < 8; ++v) {
      C[(size_t)(rowBase + v) * Nout + col]      = c0[v] + bv;
      C[(size_t)(rowBase + v + 16) * Nout + col] = c1[v] + bv;
    }
  }
}

// ---------------- LSTM pointwise: gates[32,4096] -> h, c ----------------
__global__ void lstm_pointwise_kernel(const float* __restrict__ gates,
                                      float* __restrict__ h,
                                      float* __restrict__ c)
{
  const int idx = blockIdx.x * blockDim.x + threadIdx.x;
  if (idx >= BB * DDEC_) return;
  const int b = idx >> 10, j = idx & (DDEC_ - 1);
  const float* g = gates + (size_t)b * NGATE;
  const float gi = g[j];
  const float gf = g[DDEC_ + j];
  const float gg = g[2 * DDEC_ + j];
  const float go = g[3 * DDEC_ + j];
  const float si = 1.0f / (1.0f + __expf(-gi));
  const float sf = 1.0f / (1.0f + __expf(-gf));
  const float so = 1.0f / (1.0f + __expf(-go));
  const float cn = sf * c[idx] + si * tanhf(gg);
  c[idx] = cn;
  h[idx] = so * tanhf(cn);
}

// ---------------- Attention: one block per batch, 256 threads ----------------
__global__ void attention_kernel(const float* __restrict__ enc,      // [32,512,1024]
                                 const float* __restrict__ enc_mask, // [32,512]
                                 const float* __restrict__ dec,      // [32,1024]
                                 float* __restrict__ attn)           // [32,1024]
{
  __shared__ float p[TENC];
  __shared__ float red[8];
  const int b   = blockIdx.x;
  const int tid = threadIdx.x;
  const int wid = tid >> 5, lid = tid & 31;
  const float* encb = enc + (size_t)b * TENC * DENC_;
  const float* db   = dec + (size_t)b * DENC_;

  // raw logits
  for (int t = tid; t < TENC; t += 256) {
    const float* row = encb + (size_t)t * DENC_;
    float s = 0.0f;
    for (int e = 0; e < DENC_; e += 4) {
      float4 r4 = *(const float4*)(row + e);
      float4 d4 = *(const float4*)(db + e);
      s += r4.x * d4.x + r4.y * d4.y + r4.z * d4.z + r4.w * d4.w;
    }
    p[t] = s - (1.0f - enc_mask[(size_t)b * TENC + t]) * NEGV;
  }
  __syncthreads();

  // block max
  float m = -3.0e38f;
  for (int t = tid; t < TENC; t += 256) m = fmaxf(m, p[t]);
  for (int off = 16; off > 0; off >>= 1) m = fmaxf(m, __shfl_down(m, off, 32));
  if (lid == 0) red[wid] = m;
  __syncthreads();
  if (tid == 0) {
    float mm = red[0];
    for (int i = 1; i < 8; ++i) mm = fmaxf(mm, red[i]);
    red[0] = mm;
  }
  __syncthreads();
  m = red[0];
  __syncthreads();

  // exp + block sum
  float s = 0.0f;
  for (int t = tid; t < TENC; t += 256) { float e = __expf(p[t] - m); p[t] = e; s += e; }
  for (int off = 16; off > 0; off >>= 1) s += __shfl_down(s, off, 32);
  if (lid == 0) red[wid] = s;
  __syncthreads();
  if (tid == 0) {
    float ss = 0.0f;
    for (int i = 0; i < 8; ++i) ss += red[i];
    red[0] = 1.0f / ss;
  }
  __syncthreads();
  const float inv = red[0];
  __syncthreads();

  // attn[e] = sum_t p[t] * enc[b,t,e]; thread tid owns e = tid + q*256
  float acc0 = 0.f, acc1 = 0.f, acc2 = 0.f, acc3 = 0.f;
  for (int t = 0; t < TENC; ++t) {
    const float pt = p[t] * inv;
    const float* row = encb + (size_t)t * DENC_;
    acc0 += pt * row[tid];
    acc1 += pt * row[tid + 256];
    acc2 += pt * row[tid + 512];
    acc3 += pt * row[tid + 768];
  }
  float* ab = attn + (size_t)b * DENC_;
  ab[tid] = acc0; ab[tid + 256] = acc1; ab[tid + 512] = acc2; ab[tid + 768] = acc3;
}

// ---------------- Pointer scores: 2 blocks per batch, 256 threads ----------------
__global__ void score_kernel(const float* __restrict__ emb,       // [32,512,512]
                             const float* __restrict__ emb_mask,  // [32,512]
                             const float* __restrict__ picked,    // [32,512]
                             const float* __restrict__ feat,      // [32,513]
                             float* __restrict__ out, int t)
{
  __shared__ float fsh[NFEAT];
  const int b    = blockIdx.x >> 1;
  const int half = blockIdx.x & 1;
  for (int d = threadIdx.x; d < NFEAT; d += 256) fsh[d] = feat[(size_t)b * NFEAT + d];
  __syncthreads();
  const int n = half * 256 + threadIdx.x;
  const float* e = emb + ((size_t)b * NN + n) * DEMB_;
  float s = 0.0f;
  for (int d = 0; d < DEMB_; d += 4) {
    float4 e4 = *(const float4*)(e + d);
    s += e4.x * fsh[d] + e4.y * fsh[d + 1] + e4.z * fsh[d + 2] + e4.w * fsh[d + 3];
  }
  s += picked[(size_t)b * NN + n] * fsh[DEMB_];
  s -= (1.0f - emb_mask[(size_t)b * NN + n]) * NEGV;
  out[((size_t)b * TDEC + t) * NN + n] = s;
}

// ---------------- Teacher-forced update: picked flag + next emb_t ----------------
__global__ void update_kernel(const float* __restrict__ emb,
                              const int* __restrict__ gt,
                              float* __restrict__ picked,
                              float* __restrict__ emb_t, int t)
{
  const int b = blockIdx.x;
  const int g = gt[(size_t)b * TDEC + t];
  if (threadIdx.x == 0) picked[(size_t)b * NN + g] = 1.0f;
  const float* src = emb + ((size_t)b * NN + g) * DEMB_;
  for (int d = threadIdx.x; d < DEMB_; d += blockDim.x)
    emb_t[(size_t)b * DEMB_ + d] = src[d];
}

// ---------------- Init: emb_t = go (broadcast), picked/h/c = 0 ----------------
__global__ void init_kernel(const float* __restrict__ go,
                            float* __restrict__ emb_t,
                            float* __restrict__ picked,
                            float* __restrict__ h,
                            float* __restrict__ c)
{
  const int idx = blockIdx.x * blockDim.x + threadIdx.x;
  if (idx < BB * DEMB_) { emb_t[idx] = go[idx & (DEMB_ - 1)]; picked[idx] = 0.0f; }
  if (idx < BB * DDEC_) { h[idx] = 0.0f; c[idx] = 0.0f; }
}

__global__ void bias_combine_kernel(const float* __restrict__ a,
                                    const float* __restrict__ b,
                                    float* __restrict__ o, int n)
{
  const int idx = blockIdx.x * blockDim.x + threadIdx.x;
  if (idx < n) o[idx] = a[idx] + b[idx];
}

// =====================================================================
extern "C" void kernel_launch(void* const* d_in, const int* in_sizes, int n_in,
                              void* d_out, int out_size, void* d_ws, size_t ws_size,
                              hipStream_t stream)
{
  const float* emb      = (const float*)d_in[0];
  const float* emb_mask = (const float*)d_in[1];
  const float* enc      = (const float*)d_in[2];
  const float* enc_mask = (const float*)d_in[3];
  const int*   gt       = (const int*)  d_in[4];
  const float* go       = (const float*)d_in[5];
  const float* W_ih     = (const float*)d_in[6];
  const float* W_hh     = (const float*)d_in[7];
  const float* b_ih     = (const float*)d_in[8];
  const float* b_hh     = (const float*)d_in[9];
  const float* W_dec    = (const float*)d_in[10];
  const float* b_dec    = (const float*)d_in[11];
  const float* W_feat   = (const float*)d_in[12];
  const float* b_feat   = (const float*)d_in[13];
  float* out = (float*)d_out;

  // tile counts
  const int NT_G = NGATE / 16;        // 256 column tiles, gates
  const int KT_G = KGATE / 32;        // 48
  const int NT_D = DENC_ / 16;        // 64, dec projection
  const int KT_D = DDEC_ / 32;        // 32
  const int NT_F = (NFEAT + 15) / 16; // 33 (513 padded to 528)
  const int KT_F = KFEAT / 32;        // 64

  // ---- workspace carve-up (256B aligned) ----
  char* ws = (char*)d_ws;
  size_t off = 0;
  auto carve = [&](size_t bytes) -> char* {
    char* p = ws + off;
    off = (off + bytes + 255) & ~(size_t)255;
    return p;
  };
  unsigned short* Bp_g = (unsigned short*)carve((size_t)NT_G * KT_G * 512 * 2);
  unsigned short* Bp_d = (unsigned short*)carve((size_t)NT_D * KT_D * 512 * 2);
  unsigned short* Bp_f = (unsigned short*)carve((size_t)NT_F * KT_F * 512 * 2);
  unsigned short* Ap_g = (unsigned short*)carve((size_t)2 * KT_G * 512 * 2);
  unsigned short* Ap_d = (unsigned short*)carve((size_t)2 * KT_D * 512 * 2);
  unsigned short* Ap_f = (unsigned short*)carve((size_t)2 * KT_F * 512 * 2);
  float* gates  = (float*)carve((size_t)BB * NGATE * 4);
  float* h      = (float*)carve((size_t)BB * DDEC_ * 4);
  float* c      = (float*)carve((size_t)BB * DDEC_ * 4);
  float* decb   = (float*)carve((size_t)BB * DENC_ * 4);
  float* attn   = (float*)carve((size_t)BB * DENC_ * 4);
  float* emb_t  = (float*)carve((size_t)BB * DEMB_ * 4);
  float* feat   = (float*)carve((size_t)BB * NFEAT * 4);
  float* picked = (float*)carve((size_t)BB * NN * 4);
  float* bias_g = (float*)carve((size_t)NGATE * 4);

  // ---- one-time per launch: weight packing, bias combine, state init ----
  pack_b_kernel<<<256, 256, 0, stream>>>(Bp_g, W_ih, DEMB_, W_hh, DDEC_, NGATE, NT_G);
  pack_b_kernel<<<128, 256, 0, stream>>>(Bp_d, W_dec, DDEC_, nullptr, 0, DENC_, NT_D);
  pack_b_kernel<<<128, 256, 0, stream>>>(Bp_f, W_feat, KFEAT, nullptr, 0, NFEAT, NT_F);
  bias_combine_kernel<<<(NGATE + 255) / 256, 256, 0, stream>>>(b_ih, b_hh, bias_g, NGATE);
  init_kernel<<<(BB * DDEC_ + 255) / 256, 256, 0, stream>>>(go, emb_t, picked, h, c);

  // ---- 50 recurrent steps ----
  for (int t = 0; t < TDEC; ++t) {
    // LSTM gates = [emb_t | h] @ [W_ih | W_hh]^T + (b_ih + b_hh)
    pack_a_kernel<<<64, 256, 0, stream>>>(Ap_g, emb_t, DEMB_, h, DDEC_);
    wmma_gemm_kernel<<<NT_G / 4, 128, 0, stream>>>(Ap_g, Bp_g, bias_g, gates, NT_G, KT_G, NGATE);
    lstm_pointwise_kernel<<<(BB * DDEC_ + 255) / 256, 256, 0, stream>>>(gates, h, c);

    // dec = h @ W_dec^T + b_dec
    pack_a_kernel<<<64, 256, 0, stream>>>(Ap_d, h, DDEC_, nullptr, 0);
    wmma_gemm_kernel<<<NT_D / 4, 128, 0, stream>>>(Ap_d, Bp_d, b_dec, decb, NT_D, KT_D, DENC_);

    // softmax attention over encoder states
    attention_kernel<<<BB, 256, 0, stream>>>(enc, enc_mask, decb, attn);

    // feat = [h | attn] @ W_feat^T + b_feat
    pack_a_kernel<<<64, 256, 0, stream>>>(Ap_f, h, DDEC_, attn, DENC_);
    wmma_gemm_kernel<<<(NT_F + 3) / 4, 128, 0, stream>>>(Ap_f, Bp_f, b_feat, feat, NT_F, KT_F, NFEAT);

    // pointer scores (uses picked from previous steps), then teacher update
    score_kernel<<<BB * 2, 256, 0, stream>>>(emb, emb_mask, picked, feat, out, t);
    update_kernel<<<BB, 128, 0, stream>>>(emb, gt, picked, emb_t, t);
  }
}
